// Graph_Net_art_31396210933892
// MI455X (gfx1250) — matrix operand order repaired and checked by reading, not compiled, not run on hardware
//
#include <hip/hip_runtime.h>
#include <hip/hip_bf16.h>

// ---------------- problem constants ----------------
#define BB 8
#define NV 2048
#define CC 64
#define DD 32
#define GD 128        // 4*D LSTM gate width
#define NITERS 32
#define MT 64         // rows per block in V-update kernel

typedef _Float16 v8h  __attribute__((ext_vector_type(8)));
typedef _Float16 v16h __attribute__((ext_vector_type(16)));
typedef float    v8f  __attribute__((ext_vector_type(8)));

__device__ __forceinline__ float sigm(float x) { return 1.0f / (1.0f + __expf(-x)); }

__device__ __forceinline__ v16h cat16(v8h lo, v8h hi) {
    return __builtin_shufflevector(lo, hi, 0,1,2,3,4,5,6,7,8,9,10,11,12,13,14,15);
}
// Build WMMA A-operand (16x32 f16 tile row slice) from two f32 8-vectors
__device__ __forceinline__ v16h make_a(v8f lo, v8f hi) {
    v8h l = __builtin_convertvector(lo, v8h);
    v8h h = __builtin_convertvector(hi, v8h);
    return cat16(l, h);
}
__device__ __forceinline__ v8f wmma16(v16h a, v16h b, v8f c) {
    return __builtin_amdgcn_wmma_f32_16x16x32_f16(false, a, false, b, (short)0, c, false, false);
}

// ---------------- init: hV/cV/hC/cC, fused LSTM-V weights (f16), fused bias ----
__global__ void k_init(const float* __restrict__ Vi_w, const float* __restrict__ Vi_b,
                       const float* __restrict__ Ci_w, const float* __restrict__ Ci_b,
                       const float* __restrict__ VWih, const float* __restrict__ VWhh,
                       const float* __restrict__ Vbih, const float* __restrict__ Vbhh,
                       float* __restrict__ hV, float* __restrict__ cV,
                       float* __restrict__ hC, float* __restrict__ cC,
                       _Float16* __restrict__ WcatT, float* __restrict__ biasV)
{
    int i = blockIdx.x * blockDim.x + threadIdx.x;       // 0 .. B*N*D-1
    if (i < BB*NV*DD) { int d = i & (DD-1); hV[i] = Vi_w[d] + Vi_b[d]; cV[i] = 0.0f; }
    if (i < BB*CC*DD) { int d = i & (DD-1); hC[i] = Ci_w[d] + Ci_b[d]; cC[i] = 0.0f; }
    if (i < GD*96) {
        int n = i / 96, k = i % 96;
        float w = (k < 64) ? VWih[n*64 + k] : VWhh[n*32 + (k-64)];
        WcatT[i] = (_Float16)w;
    }
    if (i < GD) biasV[i] = Vbih[i] + Vbhh[i];
}

// ---------------- one-time Adj f32 -> f16 conversion (L2-resident fast path) ---
__global__ __launch_bounds__(256)
void k_cvt_adj(const float* __restrict__ src, _Float16* __restrict__ dst)
{
    size_t i = ((size_t)blockIdx.x * blockDim.x + threadIdx.x) * 8;   // 8 elems/thread
    v8f v = *(const v8f*)(src + i);
    *(v8h*)(dst + i) = __builtin_convertvector(v, v8h);
}

// ---------------- batched transpose + f32->f16 cast: src[b][R][C] -> dst[b][C][R]
__global__ void k_transpose_f16(const float* __restrict__ src, _Float16* __restrict__ dst,
                                int R, int C)
{
    int idx = blockIdx.x * blockDim.x + threadIdx.x;     // over BATCH*R*C, r fastest in dst
    int rc = R * C;
    int bt  = idx / rc;
    int rem = idx - bt * rc;
    int c = rem / R;
    int r = rem - c * R;
    dst[idx] = (_Float16)src[(size_t)bt * rc + (size_t)r * C + c];
}

// ---------------- 3-layer MLP (D->D->D), ReLU after layers 1,2 -----------------
__global__ __launch_bounds__(256)
void k_mlp3(const float* __restrict__ x, float* __restrict__ y, int rows,
            const float* __restrict__ W1, const float* __restrict__ b1,
            const float* __restrict__ W2, const float* __restrict__ b2,
            const float* __restrict__ W3, const float* __restrict__ b3)
{
    __shared__ float sw[3*1024 + 3*32];
    int tid = threadIdx.x;
    for (int i = tid; i < 1024; i += 256) { sw[i] = W1[i]; sw[1024+i] = W2[i]; sw[2048+i] = W3[i]; }
    if (tid < 32) { sw[3072+tid] = b1[tid]; sw[3104+tid] = b2[tid]; sw[3136+tid] = b3[tid]; }
    __syncthreads();
    int r = blockIdx.x * 256 + tid;
    if (r >= rows) return;
    float xv[DD], t[DD];
    #pragma unroll
    for (int k = 0; k < DD; ++k) xv[k] = x[(size_t)r*DD + k];
    for (int o = 0; o < DD; ++o) {
        float a = sw[3072+o];
        #pragma unroll
        for (int k = 0; k < DD; ++k) a += sw[o*32+k] * xv[k];
        t[o] = fmaxf(a, 0.0f);
    }
    for (int o = 0; o < DD; ++o) {
        float a = sw[3104+o];
        #pragma unroll
        for (int k = 0; k < DD; ++k) a += sw[1024 + o*32+k] * t[k];
        xv[o] = fmaxf(a, 0.0f);
    }
    for (int o = 0; o < DD; ++o) {
        float a = sw[3136+o];
        #pragma unroll
        for (int k = 0; k < DD; ++k) a += sw[2048 + o*32+k] * xv[k];
        y[(size_t)r*DD + o] = a;
    }
}

// ---------------- V-side update: WMMA GEMMs + WMMA gates + LSTM pointwise ------
// AF16 = true : A-operand read directly from pre-converted f16 Adj (fast path)
// AF16 = false: A-operand converted f32->f16 on the fly (small-workspace fallback)
template<bool AF16>
__global__ __launch_bounds__(128)
void k_vupdate(const float* __restrict__ Adjf, const _Float16* __restrict__ Adjh,
               const float* __restrict__ Mvc,
               const _Float16* __restrict__ hVT,        // [b][D][N] f16
               const _Float16* __restrict__ mlpCoutT,   // [b][D][C] f16
               const float* __restrict__ hV_cur, float* __restrict__ hV_next,
               float* __restrict__ cV,
               const _Float16* __restrict__ WcatT,      // [128][96] f16
               const float* __restrict__ biasV)         // [128]
{
    __shared__ _Float16 sX[MT][96];      // [row][k] : adjh(0:32) | vmsg(32:64) | hV(64:96)
    __shared__ float    sG[MT][GD];      // gates

    const int tid  = threadIdx.x;
    const int wave = tid >> 5;           // 0..3
    const int lane = tid & 31;
    const int lr   = lane & 15;          // lane row/col within 16
    const int lh   = lane >> 4;          // lane half (0/1)
    const int b    = blockIdx.x >> 5;    // N/MT = 32 tiles per batch
    const int tile = blockIdx.x & 31;
    const int row0 = tile * MT + wave * 16;   // wave's 16-row group (global n)

    // ---- stage 1a: adjh = Adj[b][rows] @ hV[b]   (K = 2048, f16 WMMA) ----
    v8f acc0 = {}, acc1 = {};            // d = 0..15, 16..31
    const float*    Arowf = Adjf + ((size_t)b*NV + row0 + lr) * NV;
    const _Float16* Arowh = Adjh + ((size_t)b*NV + row0 + lr) * NV;
    const _Float16* Bp    = hVT  + (size_t)b * DD * NV;
    for (int kc = 0; kc < NV; kc += 32) {
        v16h A;
        if constexpr (AF16) {
            v8h alo = *(const v8h*)(Arowh + kc + lh*16);
            v8h ahi = *(const v8h*)(Arowh + kc + lh*16 + 8);
            A = cat16(alo, ahi);
        } else {
            v8f a_lo = *(const v8f*)(Arowf + kc + lh*16);
            v8f a_hi = *(const v8f*)(Arowf + kc + lh*16 + 8);
            A = make_a(a_lo, a_hi);
        }
        v16h B0 = *(const v16h*)(Bp + (size_t)(lr)    * NV + kc + lh*16);
        v16h B1 = *(const v16h*)(Bp + (size_t)(16+lr) * NV + kc + lh*16);
        acc0 = wmma16(A, B0, acc0);
        acc1 = wmma16(A, B1, acc1);
    }

    // ---- stage 1b: vmsg = Mvc[b][rows] @ mlp3(hC)[b]   (K = 64) ----
    v8f vm0 = {}, vm1 = {};
    const float*    Mrow = Mvc      + ((size_t)b*NV + row0 + lr) * CC;
    const _Float16* Cp   = mlpCoutT + (size_t)b * DD * CC;
    #pragma unroll
    for (int kc = 0; kc < CC; kc += 32) {
        v8f a_lo = *(const v8f*)(Mrow + kc + lh*16);
        v8f a_hi = *(const v8f*)(Mrow + kc + lh*16 + 8);
        v16h A  = make_a(a_lo, a_hi);
        v16h B0 = *(const v16h*)(Cp + (size_t)(lr)    * CC + kc + lh*16);
        v16h B1 = *(const v16h*)(Cp + (size_t)(16+lr) * CC + kc + lh*16);
        vm0 = wmma16(A, B0, vm0);
        vm1 = wmma16(A, B1, vm1);
    }

    // ---- stash X = [adjh | vmsg | hV] as f16 in LDS ----
    #pragma unroll
    for (int g = 0; g < 8; ++g) {
        int r = wave*16 + lh*8 + g;      // C/D layout: row = vgpr + 8*(lane/16)
        sX[r][lr]      = (_Float16)acc0[g];
        sX[r][16 + lr] = (_Float16)acc1[g];
        sX[r][32 + lr] = (_Float16)vm0[g];
        sX[r][48 + lr] = (_Float16)vm1[g];
    }
    for (int i = tid; i < MT*DD; i += 128) {
        int r = i >> 5, d = i & 31;
        sX[r][64 + d] = (_Float16)hV_cur[((size_t)b*NV + tile*MT + r)*DD + d];
    }
    __syncthreads();

    // ---- stage 2: gates = X(16x96) @ WcatT'(96x128) + bias, per wave ----
    const int rbase = wave * 16;
    #pragma unroll
    for (int nt = 0; nt < 8; ++nt) {
        int n0 = nt * 16;
        float bv = biasV[n0 + lr];
        v8f g = { bv, bv, bv, bv, bv, bv, bv, bv };
        #pragma unroll
        for (int kc = 0; kc < 96; kc += 32) {
            const _Float16* xp = &sX[rbase + lr][0];
            v8h alo = *(const v8h*)(xp + kc + lh*16);
            v8h ahi = *(const v8h*)(xp + kc + lh*16 + 8);
            v16h A  = cat16(alo, ahi);
            v16h Bv = *(const v16h*)(WcatT + (size_t)(n0 + lr)*96 + kc + lh*16);
            g = wmma16(A, Bv, g);
        }
        #pragma unroll
        for (int q = 0; q < 8; ++q)
            sG[rbase + lh*8 + q][n0 + lr] = g[q];
    }
    __syncthreads();

    // ---- stage 3: LSTM pointwise (gate order i,f,g,o) ----
    for (int i = tid; i < MT*DD; i += 128) {
        int r = i >> 5, d = i & 31;
        size_t gi = ((size_t)b*NV + tile*MT + r)*DD + d;
        float g_i = sG[r][d], g_f = sG[r][32+d], g_g = sG[r][64+d], g_o = sG[r][96+d];
        float c_old = cV[gi];
        float c_new = sigm(g_f) * c_old + sigm(g_i) * tanhf(g_g);
        cV[gi]      = c_new;
        hV_next[gi] = sigm(g_o) * tanhf(c_new);
    }
}

// ---------------- C-side update: c_msg reduction + C-LSTM (VALU, tiny) ---------
__global__ __launch_bounds__(256)
void k_cupdate(const float* __restrict__ Mvc, const float* __restrict__ mlpVout,
               const float* __restrict__ hC_cur, float* __restrict__ hC_next,
               float* __restrict__ cC,
               const float* __restrict__ CWih, const float* __restrict__ CWhh,
               const float* __restrict__ Cbih, const float* __restrict__ Cbhh)
{
    const int b = blockIdx.x, tid = threadIdx.x;
    __shared__ float s_m[32*CC];         // Mvc chunk   [n][c]  8KB
    __shared__ float s_v[32*DD];         // mlpV chunk  [n][d]  4KB
    __shared__ float s_cmsg[CC*DD];      //                     8KB
    __shared__ float s_hc[CC*DD];        //                     8KB
    __shared__ float s_w[2*GD*DD];       // CWih | CWhh        32KB

    float acc[8];
    #pragma unroll
    for (int j = 0; j < 8; ++j) acc[j] = 0.0f;

    for (int ch = 0; ch < NV/32; ++ch) {
        int nb = ch * 32;
        __syncthreads();
        for (int i = tid; i < 32*CC; i += 256)
            s_m[i] = Mvc[((size_t)b*NV + nb + (i>>6))*CC + (i & 63)];
        for (int i = tid; i < 32*DD; i += 256)
            s_v[i] = mlpVout[((size_t)b*NV + nb + (i>>5))*DD + (i & 31)];
        __syncthreads();
        #pragma unroll
        for (int j = 0; j < 8; ++j) {
            int o = tid + j*256; int c = o >> 5, d = o & 31;
            float a = acc[j];
            for (int n = 0; n < 32; ++n) a += s_m[n*CC + c] * s_v[n*DD + d];
            acc[j] = a;
        }
    }
    #pragma unroll
    for (int j = 0; j < 8; ++j) s_cmsg[tid + j*256] = acc[j];
    for (int i = tid; i < CC*DD; i += 256) s_hc[i] = hC_cur[(size_t)b*CC*DD + i];
    for (int i = tid; i < GD*DD; i += 256) { s_w[i] = CWih[i]; s_w[GD*DD + i] = CWhh[i]; }
    __syncthreads();

    #pragma unroll
    for (int j = 0; j < 8; ++j) {
        int o = tid + j*256; int c = o >> 5, d = o & 31;
        float g4[4];
        #pragma unroll
        for (int q = 0; q < 4; ++q) {
            int n = q*DD + d;
            float a = Cbih[n] + Cbhh[n];
            const float* wih = &s_w[n*DD];
            const float* whh = &s_w[GD*DD + n*DD];
            for (int k = 0; k < DD; ++k)
                a += wih[k]*s_cmsg[c*DD+k] + whh[k]*s_hc[c*DD+k];
            g4[q] = a;
        }
        size_t gi = (size_t)b*CC*DD + o;
        float c_old = cC[gi];
        float c_new = sigm(g4[1]) * c_old + sigm(g4[0]) * tanhf(g4[2]);
        cC[gi]      = c_new;
        hC_next[gi] = sigm(g4[3]) * tanhf(c_new);
    }
}

// ---------------- vote head: 3 linears (no activation) + sigmoid ---------------
__global__ __launch_bounds__(256)
void k_vote(const float* __restrict__ hV, float* __restrict__ out,
            const float* __restrict__ W1, const float* __restrict__ b1,
            const float* __restrict__ W2, const float* __restrict__ b2,
            const float* __restrict__ W3, const float* __restrict__ b3)
{
    __shared__ float sw[2*1024 + 32 + 2*32 + 1];
    int tid = threadIdx.x;
    for (int i = tid; i < 1024; i += 256) { sw[i] = W1[i]; sw[1024+i] = W2[i]; }
    if (tid < 32) { sw[2048+tid] = W3[tid]; sw[2080+tid] = b1[tid]; sw[2112+tid] = b2[tid]; }
    if (tid == 0) sw[2144] = b3[0];
    __syncthreads();
    int r = blockIdx.x * 256 + tid;
    if (r >= BB*NV) return;
    float xv[DD], t[DD];
    #pragma unroll
    for (int k = 0; k < DD; ++k) xv[k] = hV[(size_t)r*DD + k];
    for (int o = 0; o < DD; ++o) {
        float a = sw[2080+o];
        #pragma unroll
        for (int k = 0; k < DD; ++k) a += sw[o*32+k] * xv[k];
        t[o] = a;
    }
    float s = sw[2144];
    for (int o = 0; o < DD; ++o) {
        float a = sw[2112+o];
        #pragma unroll
        for (int k = 0; k < DD; ++k) a += sw[1024 + o*32+k] * t[k];
        s += sw[2048+o] * a;
    }
    out[r] = sigm(s);
}

// ---------------- host side -----------------------------------------------------
static inline size_t align256(size_t x) { return (x + 255) & ~(size_t)255; }

extern "C" void kernel_launch(void* const* d_in, const int* in_sizes, int n_in,
                              void* d_out, int out_size, void* d_ws, size_t ws_size,
                              hipStream_t stream) {
    const float* Adj  = (const float*)d_in[0];
    const float* Mvc  = (const float*)d_in[1];
    // d_in[2]=nvert, d_in[3]=ncol : full graphs, unused
    const float* Vi_w = (const float*)d_in[4];
    const float* Vi_b = (const float*)d_in[5];
    const float* Ci_w = (const float*)d_in[6];
    const float* Ci_b = (const float*)d_in[7];
    const float* VmW1 = (const float*)d_in[8];  const float* Vmb1 = (const float*)d_in[9];
    const float* VmW2 = (const float*)d_in[10]; const float* Vmb2 = (const float*)d_in[11];
    const float* VmW3 = (const float*)d_in[12]; const float* Vmb3 = (const float*)d_in[13];
    const float* CmW1 = (const float*)d_in[14]; const float* Cmb1 = (const float*)d_in[15];
    const float* CmW2 = (const float*)d_in[16]; const float* Cmb2 = (const float*)d_in[17];
    const float* CmW3 = (const float*)d_in[18]; const float* Cmb3 = (const float*)d_in[19];
    const float* VoW1 = (const float*)d_in[20]; const float* Vob1 = (const float*)d_in[21];
    const float* VoW2 = (const float*)d_in[22]; const float* Vob2 = (const float*)d_in[23];
    const float* VoW3 = (const float*)d_in[24]; const float* Vob3 = (const float*)d_in[25];
    const float* VWih = (const float*)d_in[26]; const float* VWhh = (const float*)d_in[27];
    const float* Vbih = (const float*)d_in[28]; const float* Vbhh = (const float*)d_in[29];
    const float* CWih = (const float*)d_in[30]; const float* CWhh = (const float*)d_in[31];
    const float* Cbih = (const float*)d_in[32]; const float* Cbhh = (const float*)d_in[33];

    // workspace layout
    char* ws = (char*)d_ws;
    size_t off = 0;
    float* hV0 = (float*)(ws + off);     off = align256(off + (size_t)BB*NV*DD*4);
    float* hV1 = (float*)(ws + off);     off = align256(off + (size_t)BB*NV*DD*4);
    float* cV  = (float*)(ws + off);     off = align256(off + (size_t)BB*NV*DD*4);
    float* hC0 = (float*)(ws + off);     off = align256(off + (size_t)BB*CC*DD*4);
    float* hC1 = (float*)(ws + off);     off = align256(off + (size_t)BB*CC*DD*4);
    float* cC  = (float*)(ws + off);     off = align256(off + (size_t)BB*CC*DD*4);
    _Float16* hVT      = (_Float16*)(ws + off); off = align256(off + (size_t)BB*DD*NV*2);
    float*    mlpVout  = (float*)(ws + off);    off = align256(off + (size_t)BB*NV*DD*4);
    float*    mlpCout  = (float*)(ws + off);    off = align256(off + (size_t)BB*CC*DD*4);
    _Float16* mlpCoutT = (_Float16*)(ws + off); off = align256(off + (size_t)BB*DD*CC*2);
    _Float16* WcatT    = (_Float16*)(ws + off); off = align256(off + (size_t)GD*96*2);
    float*    biasV    = (float*)(ws + off);    off = align256(off + (size_t)GD*4);
    // optional fast-path buffer: Adj pre-converted to f16 (67 MB, L2-resident)
    _Float16* Adj16 = (_Float16*)(ws + off);
    size_t off_f16  = off + (size_t)BB*NV*NV*2;
    const bool use_f16_adj = (off_f16 <= ws_size);
    (void)in_sizes; (void)n_in; (void)out_size;

    // init state + fused weights
    k_init<<<(BB*NV*DD + 255)/256, 256, 0, stream>>>(Vi_w, Vi_b, Ci_w, Ci_b,
                                                     VWih, VWhh, Vbih, Vbhh,
                                                     hV0, cV, hC0, cC, WcatT, biasV);
    if (use_f16_adj) {
        // 8 elems/thread: B*N*N / (256*8) blocks
        k_cvt_adj<<<(int)(((size_t)BB*NV*NV) / (256*8)), 256, 0, stream>>>(Adj, Adj16);
    }

    float* hVc = hV0; float* hVn = hV1;
    float* hCc = hC0; float* hCn = hC1;
    for (int it = 0; it < NITERS; ++it) {
        // C-side MLP and its f16 transpose (B-operand for v_msg GEMM)
        k_mlp3<<<(BB*CC + 255)/256, 256, 0, stream>>>(hCc, mlpCout, BB*CC,
                                                      CmW1, Cmb1, CmW2, Cmb2, CmW3, Cmb3);
        k_transpose_f16<<<(BB*CC*DD + 255)/256, 256, 0, stream>>>(mlpCout, mlpCoutT, CC, DD);
        // V-side MLP (input to c_msg reduction)
        k_mlp3<<<(BB*NV + 255)/256, 256, 0, stream>>>(hVc, mlpVout, BB*NV,
                                                      VmW1, Vmb1, VmW2, Vmb2, VmW3, Vmb3);
        // hV -> hVT f16 (B-operand for the big Adj GEMM)
        k_transpose_f16<<<(BB*NV*DD + 255)/256, 256, 0, stream>>>(hVc, hVT, NV, DD);
        // main WMMA kernel: adjh + vmsg GEMMs, WMMA gates, LSTM-V pointwise
        if (use_f16_adj)
            k_vupdate<true><<<BB*(NV/MT), 128, 0, stream>>>(Adj, Adj16, Mvc, hVT, mlpCoutT,
                                                            hVc, hVn, cV, WcatT, biasV);
        else
            k_vupdate<false><<<BB*(NV/MT), 128, 0, stream>>>(Adj, Adj16, Mvc, hVT, mlpCoutT,
                                                             hVc, hVn, cV, WcatT, biasV);
        // C-side: c_msg reduction + LSTM-C
        k_cupdate<<<BB, 256, 0, stream>>>(Mvc, mlpVout, hCc, hCn, cC,
                                          CWih, CWhh, Cbih, Cbhh);
        float* t;
        t = hVc; hVc = hVn; hVn = t;
        t = hCc; hCc = hCn; hCn = t;
    }
    // vote head -> sigmoid probabilities
    k_vote<<<(BB*NV + 255)/256, 256, 0, stream>>>(hVc, (float*)d_out,
                                                  VoW1, Vob1, VoW2, Vob2, VoW3, Vob3);
}
